// UVFA_text_3942779978020
// MI455X (gfx1250) — matrix operand order repaired and checked by reading, not compiled, not run on hardware
//
#include <hip/hip_runtime.h>
#include <hip/hip_bf16.h>
#include <math.h>

// ---------------------------------------------------------------------------
// UVFA_text on MI455X (gfx1250): bf16 WMMA (v_wmma_f32_16x16x32_bf16) GEMMs,
// one-hot matmuls reduced to row-gathers, object MLP hoisted out of the p-loop,
// state-MLP layer 3 folded into the final reduction.
// ---------------------------------------------------------------------------

#define B_    256
#define M2_   100
#define R_    128
#define L_    12
#define E_    64
#define SV_   32
#define OV_   32

typedef __attribute__((ext_vector_type(16))) __bf16 v16bf;
typedef __attribute__((ext_vector_type(8)))  float  v8f;

// --- fragment loaders (contiguous k = 16h + s in both A and B, wave32) ----------------
// A (16x32 bf16, MxK): lane m=lane&15; lanes>=16 carry K+16..K+31 (contiguous).
static __device__ inline v16bf load_a_frag(const __bf16* As, int stride,
                                           int m, int kbase, int koff) {
  const __bf16* p0 = As + m * stride + kbase + 2 * koff;   // K = kbase+2*koff+0..7
  const __bf16* p1 = p0 + 8;                               // K = +8..+15
  v16bf a;
#pragma unroll
  for (int i = 0; i < 8; ++i) { a[i] = p0[i]; a[8 + i] = p1[i]; }
  return a;
}
// B (32x16 bf16, KxN) from transposed weights Wt[N][K] (K contiguous):
// lane n=lane&15, K = kb + 0..15 contiguous.
static __device__ inline v16bf load_b_frag(const __bf16* Wt, int Kdim,
                                           int n, int kb) {
  const __bf16* p = Wt + n * Kdim + kb;
  v16bf b;
#pragma unroll
  for (int i = 0; i < 16; ++i) b[i] = p[i];
  return b;
}

static __device__ inline float sigm(float x) { return 1.f / (1.f + __expf(-x)); }

// --- K1: f32 -> transposed bf16 weights  dst[n*K+k] = src[k*N+n] ------------
__global__ void transpose_bf16_kernel(const float* __restrict__ src,
                                      __bf16* __restrict__ dst, int K, int N) {
  int idx = blockIdx.x * blockDim.x + threadIdx.x;
  if (idx < K * N) {
    int n = idx / K, k = idx % K;
    dst[n * K + k] = (__bf16)src[k * N + n];
  }
}

// --- K2: one-hot layer-1 as row-gather sums ---------------------------------
__global__ __launch_bounds__(128) void gather_kernel(
    const int* __restrict__ state, const int* __restrict__ obj,
    const float* __restrict__ Ws1, const float* __restrict__ bs1,
    const float* __restrict__ Wo1, const float* __restrict__ bo1,
    float* __restrict__ base_s, float* __restrict__ base_o) {
  int b = blockIdx.x, t = threadIdx.x;
  if (blockIdx.y == 0) {
    float acc = bs1[t];
    for (int c = 0; c < M2_; ++c) {
      int v = state[b * M2_ + c];
      acc += Ws1[(c * (SV_ + 1) + v) * R_ + t];
    }
    base_s[b * R_ + t] = acc;            // relu deferred (pos row added later)
  } else {
    float acc = bo1[t];
    for (int c = 0; c < M2_; ++c) {
      int v = obj[b * M2_ + c];
      acc += Wo1[(c * OV_ + v) * R_ + t];
    }
    base_o[b * R_ + t] = acc > 0.f ? acc : 0.f;
  }
}

// --- K3: object MLP tail (B rows): relu(h@Wo2+bo2)@Wo3+bo3 ------------------
__global__ __launch_bounds__(256) void obj_tail_kernel(
    const float* __restrict__ base_o, const float* __restrict__ bo2,
    const __bf16* __restrict__ Wo2t, const float* __restrict__ bo3,
    const __bf16* __restrict__ Wo3t, float* __restrict__ obj_out) {
  __shared__ __bf16 A1[16 * 128];
  __shared__ __bf16 A2[16 * 128];
  int t = threadIdx.x, blk = blockIdx.x;
  {
    int row = t >> 4, k0 = (t & 15) * 8;
    const float* src = base_o + (blk * 16 + row) * R_ + k0;
#pragma unroll
    for (int i = 0; i < 8; ++i) A1[row * 128 + k0 + i] = (__bf16)src[i];
  }
  __syncthreads();
  int wave = t >> 5, lane = t & 31;
  int m = lane & 15, koff = (lane >> 4) * 8, kb16 = (lane >> 4) * 16;
  int nn = wave * 16 + (lane & 15);
  v8f acc = {};
#pragma unroll
  for (int kk = 0; kk < 4; ++kk) {
    v16bf a = load_a_frag(A1, 128, m, kk * 32, koff);
    v16bf b = load_b_frag(Wo2t, 128, nn, kk * 32 + kb16);
    acc = __builtin_amdgcn_wmma_f32_16x16x32_bf16(false, a, false, b,
                                                  (short)0, acc, false, false);
  }
  float b2 = bo2[nn];
#pragma unroll
  for (int i = 0; i < 8; ++i) {
    int mm = i + (lane >> 4) * 8;
    float v = acc[i] + b2;
    A2[mm * 128 + nn] = (__bf16)(v > 0.f ? v : 0.f);
  }
  __syncthreads();
  v8f acc2 = {};
#pragma unroll
  for (int kk = 0; kk < 4; ++kk) {
    v16bf a = load_a_frag(A2, 128, m, kk * 32, koff);
    v16bf b = load_b_frag(Wo3t, 128, nn, kk * 32 + kb16);
    acc2 = __builtin_amdgcn_wmma_f32_16x16x32_bf16(false, a, false, b,
                                                   (short)0, acc2, false, false);
  }
  float b3 = bo3[nn];
#pragma unroll
  for (int i = 0; i < 8; ++i) {
    int mm = i + (lane >> 4) * 8;
    obj_out[(blk * 16 + mm) * R_ + nn] = acc2[i] + b3;
  }
}

// --- K4: Xpre[l][b][512] = emb[text[b,l]] @ Wih + b_lstm --------------------
__global__ __launch_bounds__(256) void xpre_kernel(
    const int* __restrict__ text, const float* __restrict__ emb,
    const __bf16* __restrict__ Wiht, const float* __restrict__ b_lstm,
    float* __restrict__ Xpre) {
  __shared__ __bf16 A[16 * 64];
  int t = threadIdx.x, blk = blockIdx.x;      // 192 blocks of 16 rows (B*L)
  {
    int row = t >> 4, k0 = (t & 15) * 4;
    int r = blk * 16 + row, b = r / L_, l = r % L_;
    int tok = text[b * L_ + l];
    const float* src = emb + tok * E_ + k0;
#pragma unroll
    for (int i = 0; i < 4; ++i) A[row * 64 + k0 + i] = (__bf16)src[i];
  }
  __syncthreads();
  int wave = t >> 5, lane = t & 31;
  int m = lane & 15, koff = (lane >> 4) * 8, kb16 = (lane >> 4) * 16;
#pragma unroll
  for (int ct = 0; ct < 4; ++ct) {
    int nn = (wave * 4 + ct) * 16 + (lane & 15);
    v8f acc = {};
#pragma unroll
    for (int kk = 0; kk < 2; ++kk) {
      v16bf a = load_a_frag(A, 64, m, kk * 32, koff);
      v16bf b = load_b_frag(Wiht, 64, nn, kk * 32 + kb16);
      acc = __builtin_amdgcn_wmma_f32_16x16x32_bf16(false, a, false, b,
                                                    (short)0, acc, false, false);
    }
    float bl = b_lstm[nn];
#pragma unroll
    for (int i = 0; i < 8; ++i) {
      int mm = i + (lane >> 4) * 8;
      int r = blk * 16 + mm, bb = r / L_, l = r % L_;
      Xpre[(l * B_ + bb) * 512 + nn] = acc[i] + bl;
    }
  }
}

// --- K5: LSTM recurrence, 16 batch rows per block, h/c in LDS ---------------
__global__ __launch_bounds__(256) void lstm_kernel(
    const float* __restrict__ Xpre, const __bf16* __restrict__ Whht,
    const float* __restrict__ obj_out, float* __restrict__ w_out) {
  __shared__ float  Hs[16 * 128];
  __shared__ float  Cs[16 * 128];
  __shared__ __bf16 Hb[16 * 128];
  __shared__ float  G[16 * 512];
  int t = threadIdx.x, blk = blockIdx.x;
  {
    int row = t >> 4, k0 = (t & 15) * 8;
#pragma unroll
    for (int i = 0; i < 8; ++i) {
      Hs[row * 128 + k0 + i] = 0.f;
      Cs[row * 128 + k0 + i] = 0.f;
      Hb[row * 128 + k0 + i] = (__bf16)0.f;
    }
  }
  __syncthreads();
  int wave = t >> 5, lane = t & 31;
  int m = lane & 15, koff = (lane >> 4) * 8, kb16 = (lane >> 4) * 16;
  for (int l = 0; l < L_; ++l) {
#pragma unroll
    for (int ct = 0; ct < 4; ++ct) {
      int nn = (wave * 4 + ct) * 16 + (lane & 15);
      v8f acc = {};
#pragma unroll
      for (int kk = 0; kk < 4; ++kk) {
        v16bf a = load_a_frag(Hb, 128, m, kk * 32, koff);
        v16bf b = load_b_frag(Whht, 128, nn, kk * 32 + kb16);
        acc = __builtin_amdgcn_wmma_f32_16x16x32_bf16(false, a, false, b,
                                                      (short)0, acc, false, false);
      }
#pragma unroll
      for (int i = 0; i < 8; ++i) {
        int mm = i + (lane >> 4) * 8;
        int r = blk * 16 + mm;
        G[mm * 512 + nn] = acc[i] + Xpre[(l * B_ + r) * 512 + nn];
      }
    }
    __syncthreads();
    {
      int row = t >> 4, u0 = (t & 15) * 8;
#pragma unroll
      for (int u = 0; u < 8; ++u) {
        int n = u0 + u;
        float ig = sigm(G[row * 512 + n]);
        float fg = sigm(G[row * 512 + 128 + n]);
        float gg = tanhf(G[row * 512 + 256 + n]);
        float og = sigm(G[row * 512 + 384 + n]);
        float c = fg * Cs[row * 128 + n] + ig * gg;
        float h = og * tanhf(c);
        Cs[row * 128 + n] = c;
        Hs[row * 128 + n] = h;
        Hb[row * 128 + n] = (__bf16)h;
      }
    }
    __syncthreads();
  }
  {
    int row = t >> 4, k0 = (t & 15) * 8;
    int r = blk * 16 + row;
#pragma unroll
    for (int i = 0; i < 8; ++i)
      w_out[r * R_ + k0 + i] = Hs[row * 128 + k0 + i] * obj_out[r * R_ + k0 + i];
  }
}

// --- K6: Wtil[b] = Ws3 @ w[b];  c0[b] = bs3 . w[b] --------------------------
__global__ __launch_bounds__(128) void combine_kernel(
    const float* __restrict__ w, const float* __restrict__ Ws3,
    const float* __restrict__ bs3, float* __restrict__ Wtil,
    float* __restrict__ c0) {
  __shared__ float wl[128];
  __shared__ float part[128];
  int b = blockIdx.x, t = threadIdx.x;
  wl[t] = w[b * R_ + t];
  __syncthreads();
  float acc = 0.f;
  for (int n = 0; n < 128; ++n) acc += Ws3[t * R_ + n] * wl[n];
  Wtil[b * R_ + t] = acc;
  part[t] = bs3[t] * wl[t];
  __syncthreads();
  if (t == 0) {
    float s = 0.f;
    for (int n = 0; n < 128; ++n) s += part[n];
    c0[b] = s;
  }
}

// --- K7: state MLP middle GEMM + folded layer-3 dot -------------------------
__global__ __launch_bounds__(256) void state_final_kernel(
    const float* __restrict__ base_s, const float* __restrict__ Ws1,
    const __bf16* __restrict__ Ws2t, const float* __restrict__ bs2,
    const float* __restrict__ Wtil, const float* __restrict__ c0,
    float* __restrict__ out) {
  __shared__ __bf16 A1[16 * 128];
  __shared__ float  H2[16 * 128];
  __shared__ float  red[16 * 16];
  int t = threadIdx.x, blk = blockIdx.x;     // 1600 blocks of 16 rows (B*M2)
  {
    int row = t >> 4, k0 = (t & 15) * 8;
    int r = blk * 16 + row, b = r / M2_, p = r % M2_;
    const float* bsrc = base_s + b * R_ + k0;
    const float* psrc = Ws1 + (p * (SV_ + 1) + SV_) * R_ + k0;
#pragma unroll
    for (int i = 0; i < 8; ++i) {
      float v = bsrc[i] + psrc[i];
      A1[row * 128 + k0 + i] = (__bf16)(v > 0.f ? v : 0.f);
    }
  }
  __syncthreads();
  int wave = t >> 5, lane = t & 31;
  int m = lane & 15, koff = (lane >> 4) * 8, kb16 = (lane >> 4) * 16;
  int nn = wave * 16 + (lane & 15);
  v8f acc = {};
#pragma unroll
  for (int kk = 0; kk < 4; ++kk) {
    v16bf a = load_a_frag(A1, 128, m, kk * 32, koff);
    v16bf b = load_b_frag(Ws2t, 128, nn, kk * 32 + kb16);
    acc = __builtin_amdgcn_wmma_f32_16x16x32_bf16(false, a, false, b,
                                                  (short)0, acc, false, false);
  }
  float b2 = bs2[nn];
#pragma unroll
  for (int i = 0; i < 8; ++i) {
    int mm = i + (lane >> 4) * 8;
    float v = acc[i] + b2;
    H2[mm * 128 + nn] = v > 0.f ? v : 0.f;
  }
  __syncthreads();
  {
    int row = t >> 4, k0 = (t & 15) * 8;
    int r = blk * 16 + row, b = r / M2_;
    float s = 0.f;
#pragma unroll
    for (int i = 0; i < 8; ++i)
      s += H2[row * 128 + k0 + i] * Wtil[b * R_ + k0 + i];
    red[row * 16 + (t & 15)] = s;
  }
  __syncthreads();
  if (t < 16) {
    int r = blk * 16 + t, b = r / M2_;
    float s = c0[b];
    for (int j = 0; j < 16; ++j) s += red[t * 16 + j];
    out[r] = s;
  }
}

// ---------------------------------------------------------------------------
extern "C" void kernel_launch(void* const* d_in, const int* in_sizes, int n_in,
                              void* d_out, int out_size, void* d_ws, size_t ws_size,
                              hipStream_t stream) {
  (void)in_sizes; (void)n_in; (void)out_size; (void)ws_size;
  const int*   state  = (const int*)d_in[0];
  const int*   obj    = (const int*)d_in[1];
  const int*   text   = (const int*)d_in[2];
  const float* Ws1    = (const float*)d_in[3];
  const float* bs1    = (const float*)d_in[4];
  const float* Ws2    = (const float*)d_in[5];
  const float* bs2    = (const float*)d_in[6];
  const float* Ws3    = (const float*)d_in[7];
  const float* bs3    = (const float*)d_in[8];
  const float* Wo1    = (const float*)d_in[9];
  const float* bo1    = (const float*)d_in[10];
  const float* Wo2    = (const float*)d_in[11];
  const float* bo2    = (const float*)d_in[12];
  const float* Wo3    = (const float*)d_in[13];
  const float* bo3    = (const float*)d_in[14];
  const float* emb    = (const float*)d_in[15];
  const float* Wih    = (const float*)d_in[16];
  const float* Whh    = (const float*)d_in[17];
  const float* b_lstm = (const float*)d_in[18];
  float* out = (float*)d_out;

  char* ws = (char*)d_ws;
  size_t off = 0;
  __bf16* Ws2t = (__bf16*)(ws + off); off += 128 * 128 * 2;
  __bf16* Wo2t = (__bf16*)(ws + off); off += 128 * 128 * 2;
  __bf16* Wo3t = (__bf16*)(ws + off); off += 128 * 128 * 2;
  __bf16* Wiht = (__bf16*)(ws + off); off += 512 * 64 * 2;
  __bf16* Whht = (__bf16*)(ws + off); off += 512 * 128 * 2;
  float* base_s  = (float*)(ws + off); off += B_ * R_ * 4;
  float* base_o  = (float*)(ws + off); off += B_ * R_ * 4;
  float* obj_out = (float*)(ws + off); off += B_ * R_ * 4;
  float* w_comb  = (float*)(ws + off); off += B_ * R_ * 4;
  float* Wtil    = (float*)(ws + off); off += B_ * R_ * 4;
  float* c0      = (float*)(ws + off); off += B_ * 4;
  float* Xpre    = (float*)(ws + off); off += (size_t)L_ * B_ * 512 * 4;

  // K1: weight transposes to bf16
  transpose_bf16_kernel<<<(128 * 128 + 255) / 256, 256, 0, stream>>>(Ws2, Ws2t, 128, 128);
  transpose_bf16_kernel<<<(128 * 128 + 255) / 256, 256, 0, stream>>>(Wo2, Wo2t, 128, 128);
  transpose_bf16_kernel<<<(128 * 128 + 255) / 256, 256, 0, stream>>>(Wo3, Wo3t, 128, 128);
  transpose_bf16_kernel<<<(64 * 512 + 255) / 256, 256, 0, stream>>>(Wih, Wiht, 64, 512);
  transpose_bf16_kernel<<<(128 * 512 + 255) / 256, 256, 0, stream>>>(Whh, Whht, 128, 512);
  // K2: one-hot layer-1 gathers
  gather_kernel<<<dim3(B_, 2), 128, 0, stream>>>(state, obj, Ws1, bs1, Wo1, bo1,
                                                 base_s, base_o);
  // K3: object MLP tail
  obj_tail_kernel<<<B_ / 16, 256, 0, stream>>>(base_o, bo2, Wo2t, bo3, Wo3t, obj_out);
  // K4: LSTM input projection
  xpre_kernel<<<(B_ * L_) / 16, 256, 0, stream>>>(text, emb, Wiht, b_lstm, Xpre);
  // K5: LSTM recurrence + combine with obj_out
  lstm_kernel<<<B_ / 16, 256, 0, stream>>>(Xpre, Whht, obj_out, w_comb);
  // K6: fold layer-3 into per-batch vector
  combine_kernel<<<B_, 128, 0, stream>>>(w_comb, Ws3, bs3, Wtil, c0);
  // K7: state MLP + final reduction -> (B,10,10)
  state_final_kernel<<<(B_ * M2_) / 16, 256, 0, stream>>>(base_s, Ws1, Ws2t, bs2,
                                                          Wtil, c0, out);
}
